// Decoder_63221918597266
// MI455X (gfx1250) — matrix operand order repaired and checked by reading, not compiled, not run on hardware
//
#include <hip/hip_runtime.h>
#include <hip/hip_bf16.h>

// ---------------- problem constants ----------------
#define E   256
#define H   8
#define DH  32
#define BB  8
#define SS  2048
#define S1  2049
#define MQ  (BB*SS)    // 16384 query rows
#define MK  (BB*S1)    // 16392 key/value rows
#define QSCALE 0.17677669529663687f   // 1/sqrt(32)

typedef __attribute__((ext_vector_type(16))) _Float16 v16h;
typedef __attribute__((ext_vector_type(8)))  _Float16 v8h;
typedef __attribute__((ext_vector_type(2)))  __fp16   v2fp;  // cvt_pkrtz result type
typedef __attribute__((ext_vector_type(8)))  float    v8f;
typedef __attribute__((ext_vector_type(4)))  unsigned v4u;
typedef __attribute__((ext_vector_type(8)))  int      v8i;
typedef __attribute__((ext_vector_type(4)))  int      v4i;
typedef unsigned short u16;

__device__ __forceinline__ u16 f2h_bits(float f) {
  union { _Float16 h; u16 s; } u;
  u.h = (_Float16)f;
  return u.s;
}

// pack two f32 -> one dword of two f16 (v_cvt_pk_rtz_f16_f32)
__device__ __forceinline__ unsigned pk2(float a, float b) {
  union { v2fp h; unsigned u; } u;
  u.h = __builtin_amdgcn_cvt_pkrtz(a, b);
  return u.u;
}

__device__ __forceinline__ v16h combine16(v8h lo, v8h hi) {
  v16h r;
#pragma unroll
  for (int i = 0; i < 8; ++i) { r[i] = lo[i]; r[i + 8] = hi[i]; }
  return r;
}

// A-operand fragment: element i <-> K = kh*8+i (i<8), 16+kh*8+(i-8) (i>=8)
__device__ __forceinline__ v16h ldfragA(const u16* row, int kh) {
  v8h lo = *(const v8h*)(row + kh * 8);
  v8h hi = *(const v8h*)(row + 16 + kh * 8);
  return combine16(lo, hi);
}

// B-operand fragment: element i <-> K = kh*8+i (i<8), 16+kh*8+(i-8) (i>=8)
__device__ __forceinline__ v16h ldfragB(const u16* row, int kh) {
  v8h lo = *(const v8h*)(row + kh * 8);
  v8h hi = *(const v8h*)(row + 16 + kh * 8);
  return combine16(lo, hi);
}

__device__ __forceinline__ v8f wmma16(v16h a, v16h b, v8f c) {
  return __builtin_amdgcn_wmma_f32_16x16x32_f16(
      false, a, false, b, (short)0, c, false, false);
}

// =====================================================================
// TDM: issue a 2-D TENSOR_LOAD_TO_LDS per CDNA5 ISA D# layout.
// data_size=2 (f16). Rows beyond tensor_dim1 read as zero (ISA OOB).
// pad: pi_code dwords-interval code, pa_code dwords-amount code.
// 6-arg builtin (clang-23): (g0 v4u, g1 v8i, g2 v4i, g3 v4i, v8i, cpol)
// =====================================================================
__device__ __forceinline__ void tdm_load_2d_f16(
    const void* gaddr, unsigned lds_off,
    unsigned dim0, unsigned dim1,     // tensor dims from origin (elements)
    unsigned tile0, unsigned tile1,   // tile dims (elements)
    unsigned long stride0,            // row stride (elements)
    unsigned pi_code, unsigned pa_code)
{
  unsigned long ga = (unsigned long)gaddr;
  v4u g0;
  g0[0] = 1u;                                            // count=1, user mode
  g0[1] = lds_off;                                       // LDS byte address
  g0[2] = (unsigned)(ga & 0xffffffffu);                  // global_addr[31:0]
  g0[3] = (unsigned)((ga >> 32) & 0x01ffffffu) | (2u << 30); // addr[56:32] | type=2
  v8i g1;
  g1[0] = (int)((1u << 16)                               // data_size = 2 bytes
              | (1u << 20)                               // pad_enable
              | (pi_code << 22) | (pa_code << 25));      // pad interval/amount
  g1[1] = (int)((dim0 & 0xffffu) << 16);                 // [15:0] atomic barrier addr=0
  g1[2] = (int)((dim0 >> 16) | ((dim1 & 0xffffu) << 16));
  g1[3] = (int)((dim1 >> 16) | (tile0 << 16));
  g1[4] = (int)(tile1 & 0xffffu);                        // tile_dim2 = 0
  g1[5] = (int)(unsigned)(stride0 & 0xffffffffu);
  g1[6] = (int)(unsigned)((stride0 >> 32) & 0xffffu);    // tensor_dim1_stride = 0
  g1[7] = 0;
  v4i z4 = {};
  v8i z8 = {};
  __builtin_amdgcn_tensor_load_to_lds(g0, g1, z4, z4, z8, 0);
}

__device__ __forceinline__ unsigned lds_offset(const void* p) {
  return (unsigned)(unsigned long long)p;   // flat LDS addr: low 32 bits = offset
}

// =====================================================================
// Pack W[K,N] f32 -> Wt[N,K] f16 (tiny, once per GEMM)
// =====================================================================
__global__ void __launch_bounds__(256)
pack_wt(const float* __restrict__ W, u16* __restrict__ Wt, int K, int N)
{
  int idx = blockIdx.x * 256 + threadIdx.x;
  if (idx >= K * N) return;
  int n = idx / K, k = idx - n * K;
  Wt[idx] = f2h_bits(W[(long)k * N + n]);
}

// =====================================================================
// GEMM: out[M,N] = act(A[M,K]f32 @ Wt[N,K]f16^T + bias); act 0/1(lrelu)
// Whole 64xK B-slice DMA'd once by TDM (padded rows, stride K+8 u16).
// block 256 = 8 waves, tile 128x64; wave: 2 A-frags x 2 B-frags = 4 WMMA.
// padcode = log2(K) - 2 (pad interval = K/2 dwords).
// =====================================================================
__global__ void __launch_bounds__(256)
gemm_wmma(const float* __restrict__ A, const u16* __restrict__ Wt,
          const float* __restrict__ bias, float* __restrict__ Cf,
          u16* __restrict__ Ch, int M, int N, int K, int act, int padcode)
{
  __shared__ __attribute__((aligned(16))) u16 As[128][40];     // [m][k]
  __shared__ __attribute__((aligned(16))) u16 Bs[64 * 264];    // [n][K+8] (max K=256)

  const int tid  = threadIdx.x;
  const int m0   = blockIdx.y * 128;
  const int n0   = blockIdx.x * 64;
  const int lane = tid & 31;
  const int wave = tid >> 5;
  const int wm   = wave & 3;   // 0..3 -> 32 rows each
  const int wn   = wave >> 2;  // 0..1 -> 32 cols each
  const int rI   = lane & 15;
  const int kh   = lane >> 4;
  const int RS   = K + 8;      // LDS row stride (u16) after TDM padding

  // ---- one TDM load of the block's whole B slice (zero-fills rows >= N) ----
  if (wave == 0) {
    tdm_load_2d_f16(Wt + (long)n0 * K, lds_offset(&Bs[0]),
                    (unsigned)K, (unsigned)(N - n0), (unsigned)K, 64u,
                    (unsigned long)K, (unsigned)padcode, 3u);
  }

  v8f acc[4] = {};             // [mi*2+ni]

  for (int kk = 0; kk < K; kk += 32) {
    // ---- stage A panel (128 x 32): 2 slots/thread, branch-free ----
#pragma unroll
    for (int it = 0; it < 2; ++it) {
      int slot = tid + it * 256;            // 0..511
      int row  = slot >> 2;
      int kg   = (slot & 3) * 8;
      int gm   = m0 + row;
      int gmc  = (gm < M) ? gm : (M - 1);   // clamp; edge rows never stored
      const float4* ap = (const float4*)(A + (long)gmc * K + kk + kg);
      if (kk + 32 < K) __builtin_prefetch(ap + 8, 0, 1);
      float4 lo = ap[0], hi = ap[1];
      uint4 p;
      p.x = pk2(lo.x, lo.y); p.y = pk2(lo.z, lo.w);
      p.z = pk2(hi.x, hi.y); p.w = pk2(hi.z, hi.w);
      *(uint4*)&As[row][kg] = p;
    }
    if (kk == 0 && wave == 0) __builtin_amdgcn_s_wait_tensorcnt(0);
    __syncthreads();

    v16h a0 = ldfragA(&As[wm * 32 + rI][0], kh);
    v16h a1 = ldfragA(&As[wm * 32 + 16 + rI][0], kh);
    v16h b0 = ldfragB(&Bs[(wn * 32 + rI) * RS + kk], kh);
    v16h b1 = ldfragB(&Bs[(wn * 32 + 16 + rI) * RS + kk], kh);
    acc[0] = wmma16(a0, b0, acc[0]);
    acc[1] = wmma16(a0, b1, acc[1]);
    acc[2] = wmma16(a1, b0, acc[2]);
    acc[3] = wmma16(a1, b1, acc[3]);
    __syncthreads();
  }

#pragma unroll
  for (int mi = 0; mi < 2; ++mi) {
#pragma unroll
    for (int j = 0; j < 8; ++j) {
      int gm = m0 + wm * 32 + mi * 16 + j + 8 * kh;
      if (gm >= M) continue;
#pragma unroll
      for (int ni = 0; ni < 2; ++ni) {
        int c = n0 + wn * 32 + ni * 16 + rI;
        if (c >= N) continue;
        float v = acc[mi * 2 + ni][j] + bias[c];
        if (act) v = (v >= 0.f) ? v : 0.2f * v;
        if (Ch) Ch[(long)gm * N + c] = f2h_bits(v);
        else    Cf[(long)gm * N + c] = v;
      }
    }
  }
}

// =====================================================================
// Flash attention, f16 Q/K/V in HBM. grid (S/128, H, B), block 256.
// K tile DMA'd per chunk by TDM (auto zero past S1); V staged transposed
// + key-permuted (s = 2*(k%16)+(k/16)) by all 256 threads.
// =====================================================================
__global__ void __launch_bounds__(256)
attn_flash(const u16* __restrict__ Qh, const u16* __restrict__ Kh,
           const u16* __restrict__ Vh, const float* __restrict__ mask,
           float* __restrict__ O)
{
  __shared__ __attribute__((aligned(16))) u16 Ks[32][40];     // [key][dh], TDM pad
  __shared__ __attribute__((aligned(16))) u16 Vs[32][40];     // [dh][s]
  __shared__ __attribute__((aligned(16))) u16 Ps[8][16][40];  // per-wave [row][s]

  const int tid  = threadIdx.x;
  const int lane = tid & 31;
  const int wave = tid >> 5;
  const int rI   = lane & 15;
  const int kh   = lane >> 4;
  const int b    = blockIdx.z;
  const int h    = blockIdx.y;
  const int q0   = blockIdx.x * 128 + wave * 16;
  const unsigned ldsK = lds_offset(&Ks[0][0]);

  // ---- Q fragment: two 16B f16 loads, scale by 1/sqrt(dh) ----
  v16h qa;
  {
    const u16* qrow = Qh + ((long)(b * SS + q0 + rI)) * E + h * DH;
    v8h lo = *(const v8h*)(qrow + kh * 8);
    v8h hi = *(const v8h*)(qrow + 16 + kh * 8);
    qa = combine16(lo, hi) * (_Float16)QSCALE;
  }

  float mx[8], ls[8];
  v8f o0 = {}, o1 = {};
#pragma unroll
  for (int j = 0; j < 8; ++j) { mx[j] = -3.0e38f; ls[j] = 0.f; }

  const int nchunks = (S1 + 31) / 32;  // 65
  for (int c = 0; c < nchunks; ++c) {
    const int key0 = c * 32;
    // ---- K tile 32x32 f16 via TDM (rows >= S1 zero-filled) ----
    if (wave == 0) {
      tdm_load_2d_f16(Kh + ((long)(b * S1 + key0)) * E + h * DH, ldsK,
                      32u, (unsigned)(S1 - key0), 32u, 32u,
                      (unsigned long)E, 3u, 3u);
    }
    // ---- V staged transposed+permuted [dh][s]: 4 elems/thread ----
    {
      int dh = tid & 31;
      int sg = (tid >> 5) * 4;
      u16 tmp[4];
#pragma unroll
      for (int j = 0; j < 4; ++j) {
        int s  = sg + j;
        int k  = ((s & 1) << 4) | (s >> 1);
        int gk = key0 + k;
        int gkc = (gk < S1) ? gk : (S1 - 1);  // padded keys get P==0
        tmp[j] = Vh[((long)(b * S1 + gkc)) * E + h * DH + dh];
      }
      uint2 p;
      p.x = (unsigned)tmp[0] | ((unsigned)tmp[1] << 16);
      p.y = (unsigned)tmp[2] | ((unsigned)tmp[3] << 16);
      *(uint2*)&Vs[dh][sg] = p;
    }
    if (wave == 0) __builtin_amdgcn_s_wait_tensorcnt(0);
    __syncthreads();

    // ---- scores: two 16x16 tiles, K-dim = dh = 32 ----
    v8f s0 = {}, s1 = {};
    {
      v16h kb0 = ldfragB(&Ks[rI][0], kh);
      v16h kb1 = ldfragB(&Ks[16 + rI][0], kh);
      s0 = wmma16(qa, kb0, s0);
      s1 = wmma16(qa, kb1, s1);
    }

    const int key_a = key0 + rI;
    const int key_b = key0 + 16 + rI;
    const float maskA = (key_a < S1) ? mask[b * S1 + key_a] * -1.0e9f : 0.f;
    const float maskB = (key_b < S1) ? mask[b * S1 + key_b] * -1.0e9f : 0.f;

    // ---- online softmax per row (row = j + 8*kh) ----
#pragma unroll
    for (int j = 0; j < 8; ++j) {
      float sa = (key_a < S1) ? (s0[j] + maskA) : -3.0e38f;
      float sb = (key_b < S1) ? (s1[j] + maskB) : -3.0e38f;
      float t = fmaxf(sa, sb);
#pragma unroll
      for (int off = 1; off < 16; off <<= 1) t = fmaxf(t, __shfl_xor(t, off, 16));
      float mn   = fmaxf(mx[j], t);
      float corr = __expf(mx[j] - mn);
      float pa   = __expf(sa - mn);
      float pb   = __expf(sb - mn);
      float psum = pa + pb;
#pragma unroll
      for (int off = 1; off < 16; off <<= 1) psum += __shfl_xor(psum, off, 16);
      ls[j] = ls[j] * corr + psum;
      o0[j] *= corr;
      o1[j] *= corr;
      mx[j] = mn;
      // packed P spill: cols s=2*rI (k=rI), s=2*rI+1 (k=16+rI)
      *(unsigned*)&Ps[wave][j + 8 * kh][2 * rI] = pk2(pa, pb);
    }
    // wave-private P: DS ops from one wave are in-order; no barrier needed

    // ---- O += P(16x32) @ V(32x32), permuted-k layout on both sides ----
    {
      v16h pf  = ldfragA(&Ps[wave][rI][0], kh);
      v16h vb0 = ldfragB(&Vs[rI][0], kh);
      v16h vb1 = ldfragB(&Vs[16 + rI][0], kh);
      o0 = wmma16(pf, vb0, o0);
      o1 = wmma16(pf, vb1, o1);
    }
    __syncthreads();
  }

  // ---- epilogue ----
#pragma unroll
  for (int j = 0; j < 8; ++j) {
    int qr = q0 + j + 8 * kh;
    float inv = 1.f / ls[j];
    float* op = O + ((long)(b * SS + qr)) * E + h * DH;
    op[rI]      = o0[j] * inv;
    op[16 + rI] = o1[j] * inv;
  }
}

// =====================================================================
// LayerNorm(A + R)*g + b, optional lrelu. One wave per 256-wide row.
// =====================================================================
__global__ void __launch_bounds__(256)
ln_residual(const float* __restrict__ A, const float* __restrict__ R,
            const float* __restrict__ g, const float* __restrict__ be,
            float* __restrict__ Out, int M, int resmode, int act)
{
  const int lane = threadIdx.x & 31;
  const int wave = threadIdx.x >> 5;
  const int row  = blockIdx.x * 8 + wave;
  if (row >= M) return;
  long rrow = row;
  if (resmode == 1) {
    int bb = row / SS, ss = row % SS;
    rrow = (long)bb * S1 + ss + 1;
  }
  const float* ap = A + (long)row * E;
  const float* rp = R + rrow * E;
  const int c0 = lane * 8;

  float4 a0 = *(const float4*)(ap + c0);
  float4 a1 = *(const float4*)(ap + c0 + 4);
  float4 r0 = *(const float4*)(rp + c0);
  float4 r1 = *(const float4*)(rp + c0 + 4);
  float x[8] = { a0.x + r0.x, a0.y + r0.y, a0.z + r0.z, a0.w + r0.w,
                 a1.x + r1.x, a1.y + r1.y, a1.z + r1.z, a1.w + r1.w };
  float s = 0.f, s2 = 0.f;
#pragma unroll
  for (int i = 0; i < 8; ++i) { s += x[i]; s2 += x[i] * x[i]; }
#pragma unroll
  for (int off = 1; off < 32; off <<= 1) {
    s  += __shfl_xor(s,  off, 32);
    s2 += __shfl_xor(s2, off, 32);
  }
  float mean = s * (1.f / E);
  float var  = fmaxf(s2 * (1.f / E) - mean * mean, 0.f);
  float rs   = rsqrtf(var + 1e-6f);
  float4 g0 = *(const float4*)(g + c0),  g1 = *(const float4*)(g + c0 + 4);
  float4 b0 = *(const float4*)(be + c0), b1 = *(const float4*)(be + c0 + 4);
  float gg[8] = { g0.x, g0.y, g0.z, g0.w, g1.x, g1.y, g1.z, g1.w };
  float bb_[8] = { b0.x, b0.y, b0.z, b0.w, b1.x, b1.y, b1.z, b1.w };
#pragma unroll
  for (int i = 0; i < 8; ++i) {
    float v = (x[i] - mean) * rs * gg[i] + bb_[i];
    if (act) v = (v >= 0.f) ? v : 0.2f * v;
    Out[(long)row * E + c0 + i] = v;
  }
}

// q1 = y[:,:,None] @ wq(1,E) + bq  -> elementwise, f16 out
__global__ void __launch_bounds__(256)
qproj1(const float* __restrict__ y, const float* __restrict__ wq,
       const float* __restrict__ bq, u16* __restrict__ Qo)
{
  long i = (long)blockIdx.x * blockDim.x + threadIdx.x;
  if (i >= (long)MQ * E) return;
  long r = i >> 8;
  int  e = (int)(i & 255);
  Qo[i] = f2h_bits(y[r] * wq[e] + bq[e]);
}

// =====================================================================
extern "C" void kernel_launch(void* const* d_in, const int* in_sizes, int n_in,
                              void* d_out, int out_size, void* d_ws, size_t ws_size,
                              hipStream_t stream)
{
  (void)in_sizes; (void)out_size; (void)ws_size;
  const float* x  = (const float*)d_in[0];
  const float* y  = (const float*)d_in[1];
  const float* xm = (const float*)d_in[2];
  const float* P[42];
  for (int i = 0; i < 42 && (3 + i) < n_in; ++i) P[i] = (const float*)d_in[3 + i];
  // P[0..7] mha1: wq bq wk bk wv bv wo bo ; P[8..15] mha2 ; P[16..23] mha3
  // P[24..31] ln1_g..ln4_b ; P[32..41] a1_w..a5_b

  // ---- workspace: fp32 activations, then f16 Q/K/V + packed weights ----
  float* ws = (float*)d_ws;
  size_t off = 0;
  float* ab  = ws + off; off += (size_t)MQ * E;
  float* cur = ws + off; off += (size_t)MQ * E;
  float* t1  = ws + off; off += (size_t)MQ * E;
  float* t2  = ws + off; off += (size_t)MQ * E;
  u16* hbase = (u16*)(ws + off);
  size_t hoff = 0;
  u16* qh  = hbase + hoff; hoff += (size_t)MQ * E;
  u16* khb = hbase + hoff; hoff += (size_t)MK * E;
  u16* vhb = hbase + hoff; hoff += (size_t)MK * E;
  u16* wt  = hbase + hoff; hoff += (size_t)E * E;

  const dim3 blk(256);
  auto padc = [](int K) { return (K == 256) ? 6 : (K == 128) ? 5 : 3; };
  auto gemmF = [&](const float* Ain, const float* W, const float* bias, float* Cf,
                   int M, int N, int K, int act) {
    pack_wt<<<dim3((unsigned)((K * N + 255) / 256)), blk, 0, stream>>>(W, wt, K, N);
    gemm_wmma<<<dim3((N + 63) / 64, (M + 127) / 128), blk, 0, stream>>>(
        Ain, wt, bias, Cf, nullptr, M, N, K, act, padc(K));
  };
  auto gemmH = [&](const float* Ain, const float* W, const float* bias, u16* Chh,
                   int M, int N, int K) {
    pack_wt<<<dim3((unsigned)((K * N + 255) / 256)), blk, 0, stream>>>(W, wt, K, N);
    gemm_wmma<<<dim3((N + 63) / 64, (M + 127) / 128), blk, 0, stream>>>(
        Ain, wt, bias, nullptr, Chh, M, N, K, 0, padc(K));
  };
  const dim3 agrid(SS / 128, H, BB);
  const dim3 lgrid(MQ / 8);

  // ---- MHA 1 ----
  {
    long tot = (long)MQ * E;
    qproj1<<<dim3((unsigned)((tot + 255) / 256)), blk, 0, stream>>>(y, P[0], P[1], qh);
  }
  gemmH(x, P[2], P[3], khb, MK, E, E);
  gemmH(x, P[4], P[5], vhb, MK, E, E);
  attn_flash<<<agrid, blk, 0, stream>>>(qh, khb, vhb, xm, ab);
  gemmF(ab, P[6], P[7], t1, MQ, E, E, 0);
  ln_residual<<<lgrid, blk, 0, stream>>>(t1, x, P[24], P[25], cur, MQ, 1, 0);

  // ---- MHA 2 ----
  gemmH(cur, P[8], P[9], qh, MQ, E, E);
  gemmH(x, P[10], P[11], khb, MK, E, E);
  gemmH(x, P[12], P[13], vhb, MK, E, E);
  attn_flash<<<agrid, blk, 0, stream>>>(qh, khb, vhb, xm, ab);
  gemmF(ab, P[14], P[15], t1, MQ, E, E, 0);
  ln_residual<<<lgrid, blk, 0, stream>>>(t1, cur, P[26], P[27], t2, MQ, 0, 0);

  // ---- MHA 3 ----
  gemmH(t2, P[16], P[17], qh, MQ, E, E);
  gemmH(x, P[18], P[19], khb, MK, E, E);
  gemmH(x, P[20], P[21], vhb, MK, E, E);
  attn_flash<<<agrid, blk, 0, stream>>>(qh, khb, vhb, xm, ab);
  gemmF(ab, P[22], P[23], t1, MQ, E, E, 0);
  ln_residual<<<lgrid, blk, 0, stream>>>(t1, t2, P[28], P[29], cur, MQ, 0, 1);

  // ---- MLP head ----
  gemmF(cur, P[32], P[33], t1, MQ, 256, 256, 1);  // a1 + lrelu
  gemmF(t1,  P[34], P[35], t2, MQ, 256, 256, 0);  // a2
  ln_residual<<<lgrid, blk, 0, stream>>>(t2, cur, P[30], P[31], t1, MQ, 0, 0); // ln4
  gemmF(t1, P[36], P[37], t2, MQ, 128, 256, 1);   // a3 + lrelu
  gemmF(t2, P[38], P[39], ab, MQ, 32, 128, 1);    // a4 + lrelu
  gemmF(ab, P[40], P[41], (float*)d_out, MQ, 2, 32, 0); // a5 -> (MQ x 2)
}
